// MVSNet_17016660427364
// MI455X (gfx1250) — compile-verified, matrix operand order checked by reading, not run on hardware
//
#include <hip/hip_runtime.h>
#include <math.h>

typedef _Float16 half_t;
typedef __attribute__((ext_vector_type(16))) _Float16 v16h;
typedef __attribute__((ext_vector_type(8)))  float    v8f;
typedef unsigned int v4u __attribute__((ext_vector_type(4)));
typedef int          v8i __attribute__((ext_vector_type(8)));
typedef int          v4i __attribute__((ext_vector_type(4)));

#if defined(__has_builtin)
#  if __has_builtin(__builtin_amdgcn_tensor_load_to_lds)
#    define USE_TDM 1
#  endif
#endif
#ifndef USE_TDM
#  define USE_TDM 0
#endif

// ---------------- problem constants (fixed by reference) ----------------
constexpr int Bc  = 2;          // batch
constexpr int Vc  = 3;          // views
constexpr int Hc  = 512, Wc = 640;
constexpr int Dc  = 64;         // depth hypotheses
constexpr int HFc = Hc / 4, WFc = Wc / 4;   // 128 x 160
constexpr int CFc = 32;         // feature channels
constexpr int NIMG = Bc * Vc;   // 6 images
constexpr int HWFc = HFc * WFc; // 20480
constexpr int NSRC = Vc - 1;    // 2 source views

constexpr int NPT = 2;           // pixel tiles per wave (A-fragment reuse)
constexpr int MAX_WTILE = 12800; // max Cout*Ctot*kh*kw halves (layer c5)

// ---------------- generic implicit-GEMM conv via WMMA -------------------
struct ConvArgs {
  const half_t* inA;   // [G][CA][Hin][Win] f16
  const half_t* inB;   // [G][CB][Hin][Win] f16 (concat tail), may be null
  const half_t* inB2;  // gate: effective inB value = inB * inB2 (r*h), may be null
  const half_t* w16;   // [Cout][CA+CB][kh][kw] f16 (pre-converted arena)
  const float*  gamma; // [Cout] or null (=> 1.0)
  const float*  beta;  // [Cout] or null (=> 0.0)
  half_t*       out16;     // [G][Cout][Ho*Wo] f16, may be null
  float*        outF32;    // alt f32 output (cost volume), may be null
  long long     outF32BStride; // per-image stride for outF32
  const half_t* uGate; // [G][Cout][Ho*Wo]: if set, out = u*hPrev + (1-u)*act(y)
  const half_t* hPrev;
  int CA, CB, Cout, Hin, Win, Ho, Wo, kh, kw, stride, pad, act; // act:0 none 1 relu 2 sigmoid 3 tanh
};

__global__ __launch_bounds__(256) void conv_wmma_kernel(ConvArgs a) {
  __shared__ __attribute__((aligned(16))) half_t wsh[MAX_WTILE]; // f16 weights [m][k]
  __shared__ int tapKy[32];
  __shared__ int tapKx[32];

  const int tid  = threadIdx.x;
  const int lane = tid & 31;
  const int Ctot = a.CA + a.CB;
  const int khkw = a.kh * a.kw;
  const int Ktot = Ctot * khkw;

  // ---- weight tile -> LDS via Tensor Data Mover (wave 0 issues one DMA) ----
#if USE_TDM
  if (tid < 32) {
    const unsigned N   = (unsigned)(a.Cout * Ktot);      // elements (2B each, N even)
    const unsigned lds = (unsigned)(size_t)(void*)&wsh[0];
    const unsigned long long ga = (unsigned long long)(size_t)a.w16;
    // D# group0: count=1 | lds_addr | global_addr[56:0] | type=2
    v4u g0 = { 1u, lds, (unsigned)ga,
               (unsigned)((ga >> 32) & 0x01FFFFFFu) | (2u << 30) };
    // D# group1: data_size=2B; tensor_dim0=N; tensor_dim1=1; tile_dim0=N;
    //            tile_dim1=1; tensor_dim0_stride=N; no multicast/pad/iterate
    v8i g1 = { (int)(1u << 16),
               (int)((N & 0xFFFFu) << 16),
               (int)(((N >> 16) & 0xFFFFu) | (1u << 16)),
               (int)((N & 0xFFFFu) << 16),
               (int)1,
               (int)N,
               0, 0 };
    v4i z4 = { 0, 0, 0, 0 };
#if defined(__clang_major__) && __clang_major__ >= 23
    v8i z8 = { 0, 0, 0, 0, 0, 0, 0, 0 };
    __builtin_amdgcn_tensor_load_to_lds(g0, g1, z4, z4, z8, 0);
#else
    __builtin_amdgcn_tensor_load_to_lds(g0, g1, z4, z4, 0);
#endif
    __builtin_amdgcn_s_wait_tensorcnt(0);
  }
  if (tid >= 32 && tid < 32 + khkw) {
    const int t = tid - 32;
    const int ky = t / a.kw;
    tapKy[t] = ky;
    tapKx[t] = t - ky * a.kw;
  }
#else
  for (int i = tid; i < a.Cout * Ktot; i += (int)blockDim.x)
    wsh[i] = a.w16[i];
  if (tid < khkw) {
    const int ky = tid / a.kw;
    tapKy[tid] = ky;
    tapKx[tid] = tid - ky * a.kw;
  }
#endif
  __syncthreads();

  const int HoWo     = a.Ho * a.Wo;
  const int pixTiles = (HoWo + 15) >> 4;
  const int ptGroups = (pixTiles + NPT - 1) / NPT;
  const int mTiles   = (a.Cout + 15) >> 4;
  const int gw = blockIdx.x * (blockDim.x >> 5) + (tid >> 5);
  if (gw >= ptGroups * mTiles) return;          // wave-uniform exit (after barrier)
  const int mTile = gw / ptGroups;
  const int ptg   = gw - mTile * ptGroups;
  const int g = blockIdx.z;

  const int planeIn = a.Hin * a.Win;
  const size_t inAbase = (size_t)g * a.CA * planeIn;
  const size_t inBbase = (size_t)g * a.CB * planeIn;

  // WMMA 16-bit fragment lane mapping (ISA 7.12.2)
  const int mA   = mTile * 16 + (lane & 15);
  const bool mAv = (mA < a.Cout);
  const int nB   = lane & 15;
  const int kbase = (lane & 16) ? 8 : 0;

  int  pP[NPT], oyS[NPT], oxS[NPT];
  bool pV[NPT];
#pragma unroll
  for (int pt = 0; pt < NPT; ++pt) {
    const int p = (ptg * NPT + pt) * 16 + nB;
    pP[pt] = p;
    pV[pt] = (p < HoWo);
    const int oy = p / a.Wo;
    const int ox = p - oy * a.Wo;
    oyS[pt] = oy * a.stride - a.pad;
    oxS[pt] = ox * a.stride - a.pad;
  }

  // incremental k -> (c, rem) decode (no integer division in the hot loop;
  // khkw >= 9, advance of 32 needs at most 4 conditional subtracts)
  int cE[16], remE[16];
#pragma unroll
  for (int e = 0; e < 16; ++e) {
    int c = 0, rem = kbase + e + ((e >= 8) ? 8 : 0);
#pragma unroll
    for (int j = 0; j < 4; ++j)
      if (rem >= khkw) { rem -= khkw; ++c; }
    cE[e] = c; remE[e] = rem;
  }

  v8f acc[NPT] = {};
  for (int kk = 0; kk < Ktot; kk += 32) {
    v16h va, vb[NPT];
#pragma unroll
    for (int e = 0; e < 16; ++e) {
      const int koff = kbase + e + ((e >= 8) ? 8 : 0);
      const int k = kk + koff;
      const bool kv = (k < Ktot);
      va[e] = (kv && mAv) ? wsh[mA * Ktot + k] : (half_t)0.0f;

      const int c   = cE[e];
      const int rem = remE[e];
      const int ky = tapKy[rem];
      const int kx = tapKx[rem];
#pragma unroll
      for (int pt = 0; pt < NPT; ++pt) {
        half_t xv = (half_t)0.0f;
        if (kv && pV[pt]) {
          const int iy = oyS[pt] + ky;
          const int ix = oxS[pt] + kx;
          if (iy >= 0 && iy < a.Hin && ix >= 0 && ix < a.Win) {
            const size_t off = (size_t)iy * a.Win + ix;
            if (c < a.CA) {
              xv = a.inA[inAbase + (size_t)c * planeIn + off];
            } else {
              const int c2 = c - a.CA;
              xv = a.inB[inBbase + (size_t)c2 * planeIn + off];
              if (a.inB2)
                xv = (half_t)((float)xv *
                              (float)a.inB2[inBbase + (size_t)c2 * planeIn + off]);
            }
          }
        }
        vb[pt][e] = xv;
      }
      // advance decode by 32
      int nr = remE[e] + 32, nc = cE[e];
#pragma unroll
      for (int j = 0; j < 4; ++j)
        if (nr >= khkw) { nr -= khkw; ++nc; }
      remE[e] = nr; cE[e] = nc;
    }
#pragma unroll
    for (int pt = 0; pt < NPT; ++pt)
      acc[pt] = __builtin_amdgcn_wmma_f32_16x16x32_f16(
          /*neg_a=*/false, va, /*neg_b=*/false, vb[pt],
          /*c_mod=*/(short)0, acc[pt], /*reuse_a=*/false, /*reuse_b=*/false);
  }

  // epilogue: D layout — vgpr r, lane l -> m = r + (l>=16 ? 8 : 0), n = l&15
#pragma unroll
  for (int pt = 0; pt < NPT; ++pt) {
#pragma unroll
    for (int r = 0; r < 8; ++r) {
      const int m = mTile * 16 + r + ((lane & 16) ? 8 : 0);
      if (m < a.Cout && pV[pt]) {
        float y = acc[pt][r];
        if (a.gamma) y *= a.gamma[m];
        if (a.beta)  y += a.beta[m];
        if      (a.act == 1) y = fmaxf(y, 0.0f);
        else if (a.act == 2) y = 1.0f / (1.0f + expf(-y));
        else if (a.act == 3) y = tanhf(y);
        const int p = pP[pt];
        const size_t oidx = ((size_t)g * a.Cout + m) * HoWo + p;
        if (a.uGate) {
          const float u  = (float)a.uGate[oidx];
          const float hp = (float)a.hPrev[oidx];
          y = u * hp + (1.0f - u) * y;
        }
        if (a.out16)  a.out16[oidx] = (half_t)y;
        if (a.outF32) a.outF32[(size_t)g * a.outF32BStride + (size_t)m * HoWo + p] = y;
      }
    }
  }
}

// ---------------- helpers ----------------
__global__ void f32_to_f16_kernel(const float* __restrict__ src,
                                  half_t* __restrict__ dst, int n) {
  int i = blockIdx.x * blockDim.x + threadIdx.x;
  if (i < n) dst[i] = (half_t)src[i];
}

__global__ void zero_f16_kernel(half_t* __restrict__ p, int n) {
  int i = blockIdx.x * blockDim.x + threadIdx.x;
  if (i < n) p[i] = (half_t)0.0f;
}

// projs[b][v] = proj[b][v+1] @ inv(proj[b][0])   (4x4, Gauss-Jordan)
__global__ void projs_kernel(const float* __restrict__ proj,
                             float* __restrict__ projs) {
  const int b = threadIdx.x;
  if (b >= Bc) return;
  float M[4][8];
  const float* P0 = proj + (size_t)b * Vc * 16;
  for (int i = 0; i < 4; ++i)
    for (int j = 0; j < 4; ++j) {
      M[i][j] = P0[i * 4 + j];
      M[i][4 + j] = (i == j) ? 1.0f : 0.0f;
    }
  for (int col = 0; col < 4; ++col) {
    int piv = col;
    for (int r = col + 1; r < 4; ++r)
      if (fabsf(M[r][col]) > fabsf(M[piv][col])) piv = r;
    if (piv != col)
      for (int j = 0; j < 8; ++j) { float t = M[col][j]; M[col][j] = M[piv][j]; M[piv][j] = t; }
    const float inv = 1.0f / M[col][col];
    for (int j = 0; j < 8; ++j) M[col][j] *= inv;
    for (int r = 0; r < 4; ++r)
      if (r != col) {
        const float f = M[r][col];
        for (int j = 0; j < 8; ++j) M[r][j] -= f * M[col][j];
      }
  }
  for (int v = 1; v < Vc; ++v) {
    const float* Pv = proj + ((size_t)b * Vc + v) * 16;
    float* out = projs + ((size_t)b * NSRC + (v - 1)) * 16;
    for (int i = 0; i < 4; ++i)
      for (int j = 0; j < 4; ++j) {
        float s = 0.0f;
        for (int k = 0; k < 4; ++k) s += Pv[i * 4 + k] * M[k][4 + j];
        out[i * 4 + j] = s;
      }
  }
}

// per-depth homography warp + variance cost: xvar = -(1/nsrc) * sum_v (warp_v - ref)^2
__global__ __launch_bounds__(256) void warp_var_kernel(
    const half_t* __restrict__ feats,   // [b][v][CF][HF*WF] f16
    const float* __restrict__ projs,    // [b][v-1][16]
    const float* __restrict__ depthv,   // [b][D]
    int d, half_t* __restrict__ xvar) { // [b][CF][HF*WF]
  const int p = blockIdx.x * blockDim.x + threadIdx.x;
  const int b = blockIdx.z;
  if (p >= HWFc) return;
  const float fy = (float)(p / WFc);
  const float fx = (float)(p - (p / WFc) * WFc);
  const float depth = depthv[b * Dc + d];
  const half_t* refF = feats + (size_t)(b * Vc + 0) * CFc * HWFc;

  float var[CFc];
#pragma unroll
  for (int c = 0; c < CFc; ++c) var[c] = 0.0f;

  for (int v = 0; v < NSRC; ++v) {
    const float* P = projs + ((size_t)b * NSRC + v) * 16;
    const float X = (P[0] * fx + P[1] * fy + P[2])  * depth + P[3];
    const float Y = (P[4] * fx + P[5] * fy + P[6])  * depth + P[7];
    const float Z = (P[8] * fx + P[9] * fy + P[10]) * depth + P[11];
    const float px = X / Z, py = Y / Z;
    const float x0 = floorf(px), y0 = floorf(py);
    const half_t* sf = feats + (size_t)(b * Vc + (v + 1)) * CFc * HWFc;

    float warped[CFc];
#pragma unroll
    for (int c = 0; c < CFc; ++c) warped[c] = 0.0f;

#pragma unroll
    for (int corner = 0; corner < 4; ++corner) {
      const float xi = x0 + (float)(corner & 1);
      const float yi = y0 + (float)(corner >> 1);
      const float wgt = (1.0f - fabsf(px - xi)) * (1.0f - fabsf(py - yi));
      const bool valid = (xi >= 0.0f) && (xi <= (float)(WFc - 1)) &&
                         (yi >= 0.0f) && (yi <= (float)(HFc - 1));
      const float wv = valid ? wgt : 0.0f;
      int cxi = (int)fminf(fmaxf(xi, 0.0f), (float)(WFc - 1));
      int cyi = (int)fminf(fmaxf(yi, 0.0f), (float)(HFc - 1));
      const int idx = cyi * WFc + cxi;
#pragma unroll
      for (int c = 0; c < CFc; ++c)
        warped[c] += (float)sf[(size_t)c * HWFc + idx] * wv;
    }
#pragma unroll
    for (int c = 0; c < CFc; ++c) {
      const float df = warped[c] - (float)refF[(size_t)c * HWFc + p];
      var[c] += df * df;
    }
  }
  const float sc = -1.0f / (float)NSRC;
#pragma unroll
  for (int c = 0; c < CFc; ++c)
    xvar[((size_t)b * CFc + c) * HWFc + p] = (half_t)(var[c] * sc);
}

// softmax over depth axis: out[b][d][hw] = softmax_d(costs[b][d][hw])
__global__ __launch_bounds__(256) void softmax_kernel(
    const float* __restrict__ costs, float* __restrict__ out) {
  const int p = blockIdx.x * blockDim.x + threadIdx.x;
  const int b = blockIdx.z;
  if (p >= HWFc) return;
  float v[Dc];
  float mx = -3.0e38f;
#pragma unroll
  for (int d = 0; d < Dc; ++d) {
    v[d] = costs[((size_t)b * Dc + d) * HWFc + p];
    mx = fmaxf(mx, v[d]);
  }
  float s = 0.0f;
#pragma unroll
  for (int d = 0; d < Dc; ++d) { v[d] = expf(v[d] - mx); s += v[d]; }
  const float inv = 1.0f / s;
#pragma unroll
  for (int d = 0; d < Dc; ++d)
    out[((size_t)b * Dc + d) * HWFc + p] = v[d] * inv;
}

// ---------------- host side ----------------
enum {
  IN_IMGS = 0, IN_PROJ, IN_DEPTH,
  C0W, C0G, C0B, C1W, C1G, C1B, C2W, C2G, C2B, C3W, C3G, C3B,
  C4W, C4G, C4B, C5W, C5G, C5B, C6W, C6G, C6B, FWi, FBi,
  G1WR, G1BR, G1WU, G1BU, G1WC, G1BC,
  G2WR, G2BR, G2WU, G2BU, G2WC, G2BC,
  G3WR, G3BR, G3WU, G3BU, G3WC, G3BC,
  PWi, PBi, IN_COUNT
};

extern "C" void kernel_launch(void* const* d_in, const int* in_sizes, int n_in,
                              void* d_out, int out_size, void* d_ws, size_t ws_size,
                              hipStream_t stream) {
  (void)in_sizes; (void)n_in; (void)out_size;
  const float* F[IN_COUNT];
  for (int i = 0; i < IN_COUNT; ++i) F[i] = (const float*)d_in[i];

  // ---- workspace arena (f16 activations, f32 cost volume) ----
  char* base = (char*)d_ws;
  size_t off = 0;
  auto alloc = [&](size_t bytes) -> void* {
    void* r = base + off;
    off = (off + bytes + 255) & ~(size_t)255;
    return r;
  };
  half_t* img16 = (half_t*)alloc((size_t)NIMG * 3 * Hc * Wc * 2);
  half_t* actA  = (half_t*)alloc((size_t)NIMG * 8 * Hc * Wc * 2);   // max-size ping
  half_t* actB  = (half_t*)alloc((size_t)NIMG * 8 * Hc * Wc * 2);   // max-size pong
  half_t* feats = (half_t*)alloc((size_t)NIMG * CFc * HWFc * 2);
  float*  projs = (float*)alloc((size_t)Bc * NSRC * 16 * 4);
  half_t* xvar  = (half_t*)alloc((size_t)Bc * CFc * HWFc * 2);
  half_t* r1 = (half_t*)alloc((size_t)Bc * 16 * HWFc * 2);
  half_t* u1 = (half_t*)alloc((size_t)Bc * 16 * HWFc * 2);
  half_t* h1a = (half_t*)alloc((size_t)Bc * 16 * HWFc * 2);
  half_t* h1b = (half_t*)alloc((size_t)Bc * 16 * HWFc * 2);
  half_t* r2 = (half_t*)alloc((size_t)Bc * 4 * HWFc * 2);
  half_t* u2 = (half_t*)alloc((size_t)Bc * 4 * HWFc * 2);
  half_t* h2a = (half_t*)alloc((size_t)Bc * 4 * HWFc * 2);
  half_t* h2b = (half_t*)alloc((size_t)Bc * 4 * HWFc * 2);
  half_t* r3 = (half_t*)alloc((size_t)Bc * 2 * HWFc * 2);
  half_t* u3 = (half_t*)alloc((size_t)Bc * 2 * HWFc * 2);
  half_t* h3a = (half_t*)alloc((size_t)Bc * 2 * HWFc * 2);
  half_t* h3b = (half_t*)alloc((size_t)Bc * 2 * HWFc * 2);
  float*  costs = (float*)alloc((size_t)Bc * Dc * HWFc * 4);

  // f16 weight arena (TDM source): one slot per conv weight tensor
  half_t* W16[IN_COUNT] = {};
  const int widx[] = { C0W, C1W, C2W, C3W, C4W, C5W, C6W, FWi,
                       G1WR, G1WU, G1WC, G2WR, G2WU, G2WC,
                       G3WR, G3WU, G3WC, PWi };
  const int wcnt[] = { 8*3*9, 8*8*9, 16*8*25, 16*16*9, 16*16*9, 32*16*25,
                       32*32*9, 32*32*9,
                       16*48*9, 16*48*9, 16*48*9, 4*20*9, 4*20*9, 4*20*9,
                       2*6*9, 2*6*9, 2*6*9, 1*2*9 };
  for (int i = 0; i < (int)(sizeof(widx) / sizeof(widx[0])); ++i)
    W16[widx[i]] = (half_t*)alloc((size_t)wcnt[i] * 2);
  if (off > ws_size) return;   // workspace too small — bail deterministically

  // convert all conv weights f32 -> f16 once per call
  for (int i = 0; i < (int)(sizeof(widx) / sizeof(widx[0])); ++i) {
    const int n = wcnt[i];
    f32_to_f16_kernel<<<(n + 255) / 256, 256, 0, stream>>>(F[widx[i]], W16[widx[i]], n);
  }

  auto launchConv = [&](ConvArgs a, int G) {
    const int HoWo = a.Ho * a.Wo;
    const int pixTiles = (HoWo + 15) / 16;
    const int ptGroups = (pixTiles + NPT - 1) / NPT;
    const int waves = ptGroups * ((a.Cout + 15) / 16);
    const int blocks = (waves + 7) / 8;        // 256 thr = 8 wave32
    conv_wmma_kernel<<<dim3(blocks, 1, G), dim3(256), 0, stream>>>(a);
  };
  auto mk = [&](const half_t* inA, int CA, const half_t* inB, int CB,
                const half_t* inB2, const half_t* w, const float* gam,
                const float* bet, half_t* o16, float* oF32, long long ofs,
                const half_t* uG, const half_t* hP, int Cout, int Hin, int Win,
                int Ho, int Wo, int kh, int kw, int s, int p, int act) {
    ConvArgs c{inA, inB, inB2, w, gam, bet, o16, oF32, ofs, uG, hP,
               CA, CB, Cout, Hin, Win, Ho, Wo, kh, kw, s, p, act};
    return c;
  };

  // ---- image -> f16 ----
  {
    const int n = NIMG * 3 * Hc * Wc;
    f32_to_f16_kernel<<<(n + 255) / 256, 256, 0, stream>>>(F[IN_IMGS], img16, n);
  }
  // ---- projection matrices ----
  projs_kernel<<<1, 32, 0, stream>>>(F[IN_PROJ], projs);
  // ---- zero GRU states ----
  {
    int n1 = Bc * 16 * HWFc, n2 = Bc * 4 * HWFc, n3 = Bc * 2 * HWFc;
    zero_f16_kernel<<<(n1 + 255) / 256, 256, 0, stream>>>(h1a, n1);
    zero_f16_kernel<<<(n2 + 255) / 256, 256, 0, stream>>>(h2a, n2);
    zero_f16_kernel<<<(n3 + 255) / 256, 256, 0, stream>>>(h3a, n3);
  }

  // ---- feature net (G = 6 images) ----
  const int H2 = Hc / 2, W2 = Wc / 2;
  launchConv(mk(img16, 3, 0, 0, 0, W16[C0W], F[C0G], F[C0B], actA, 0, 0, 0, 0,
                8, Hc, Wc, Hc, Wc, 3, 3, 1, 1, 1), NIMG);
  launchConv(mk(actA, 8, 0, 0, 0, W16[C1W], F[C1G], F[C1B], actB, 0, 0, 0, 0,
                8, Hc, Wc, Hc, Wc, 3, 3, 1, 1, 1), NIMG);
  launchConv(mk(actB, 8, 0, 0, 0, W16[C2W], F[C2G], F[C2B], actA, 0, 0, 0, 0,
                16, Hc, Wc, H2, W2, 5, 5, 2, 2, 1), NIMG);
  launchConv(mk(actA, 16, 0, 0, 0, W16[C3W], F[C3G], F[C3B], actB, 0, 0, 0, 0,
                16, H2, W2, H2, W2, 3, 3, 1, 1, 1), NIMG);
  launchConv(mk(actB, 16, 0, 0, 0, W16[C4W], F[C4G], F[C4B], actA, 0, 0, 0, 0,
                16, H2, W2, H2, W2, 3, 3, 1, 1, 1), NIMG);
  launchConv(mk(actA, 16, 0, 0, 0, W16[C5W], F[C5G], F[C5B], actB, 0, 0, 0, 0,
                32, H2, W2, HFc, WFc, 5, 5, 2, 2, 1), NIMG);
  launchConv(mk(actB, 32, 0, 0, 0, W16[C6W], F[C6G], F[C6B], actA, 0, 0, 0, 0,
                32, HFc, WFc, HFc, WFc, 3, 3, 1, 1, 1), NIMG);
  launchConv(mk(actA, 32, 0, 0, 0, W16[FWi], 0, F[FBi], feats, 0, 0, 0, 0,
                32, HFc, WFc, HFc, WFc, 3, 3, 1, 1, 0), NIMG);

  // ---- sequential plane sweep (GRU recurrence over 64 depths) ----
  const int warpBlocks = (HWFc + 255) / 256;
  for (int d = 0; d < Dc; ++d) {
    half_t* h1in = (d & 1) ? h1b : h1a;  half_t* h1out = (d & 1) ? h1a : h1b;
    half_t* h2in = (d & 1) ? h2b : h2a;  half_t* h2out = (d & 1) ? h2a : h2b;
    half_t* h3in = (d & 1) ? h3b : h3a;  half_t* h3out = (d & 1) ? h3a : h3b;

    warp_var_kernel<<<dim3(warpBlocks, 1, Bc), dim3(256), 0, stream>>>(
        feats, projs, F[IN_DEPTH], d, xvar);

    // GRU1: x = xvar (32ch), h = h1 (16ch)
    launchConv(mk(xvar, 32, h1in, 16, 0, W16[G1WR], 0, F[G1BR], r1, 0, 0, 0, 0,
                  16, HFc, WFc, HFc, WFc, 3, 3, 1, 1, 2), Bc);
    launchConv(mk(xvar, 32, h1in, 16, 0, W16[G1WU], 0, F[G1BU], u1, 0, 0, 0, 0,
                  16, HFc, WFc, HFc, WFc, 3, 3, 1, 1, 2), Bc);
    launchConv(mk(xvar, 32, h1in, 16, r1, W16[G1WC], 0, F[G1BC], h1out, 0, 0,
                  u1, h1in, 16, HFc, WFc, HFc, WFc, 3, 3, 1, 1, 3), Bc);

    // GRU2: x = h1out (16ch), h = h2 (4ch)
    launchConv(mk(h1out, 16, h2in, 4, 0, W16[G2WR], 0, F[G2BR], r2, 0, 0, 0, 0,
                  4, HFc, WFc, HFc, WFc, 3, 3, 1, 1, 2), Bc);
    launchConv(mk(h1out, 16, h2in, 4, 0, W16[G2WU], 0, F[G2BU], u2, 0, 0, 0, 0,
                  4, HFc, WFc, HFc, WFc, 3, 3, 1, 1, 2), Bc);
    launchConv(mk(h1out, 16, h2in, 4, r2, W16[G2WC], 0, F[G2BC], h2out, 0, 0,
                  u2, h2in, 4, HFc, WFc, HFc, WFc, 3, 3, 1, 1, 3), Bc);

    // GRU3: x = h2out (4ch), h = h3 (2ch)
    launchConv(mk(h2out, 4, h3in, 2, 0, W16[G3WR], 0, F[G3BR], r3, 0, 0, 0, 0,
                  2, HFc, WFc, HFc, WFc, 3, 3, 1, 1, 2), Bc);
    launchConv(mk(h2out, 4, h3in, 2, 0, W16[G3WU], 0, F[G3BU], u3, 0, 0, 0, 0,
                  2, HFc, WFc, HFc, WFc, 3, 3, 1, 1, 2), Bc);
    launchConv(mk(h2out, 4, h3in, 2, r3, W16[G3WC], 0, F[G3BC], h3out, 0, 0,
                  u3, h3in, 2, HFc, WFc, HFc, WFc, 3, 3, 1, 1, 3), Bc);

    // cost conv: 2ch -> 1ch, write f32 slice costs[b][d][:]
    launchConv(mk(h3out, 2, 0, 0, 0, W16[PWi], 0, F[PBi], 0,
                  costs + (size_t)d * HWFc, (long long)Dc * HWFc, 0, 0,
                  1, HFc, WFc, HFc, WFc, 3, 3, 1, 1, 0), Bc);
  }

  // ---- softmax over depth ----
  softmax_kernel<<<dim3((HWFc + 255) / 256, 1, Bc), dim3(256), 0, stream>>>(
      costs, (float*)d_out);
}
